// DeltaVPredictor_52974126629387
// MI455X (gfx1250) — matrix operand order repaired and checked by reading, not compiled, not run on hardware
//
#include <hip/hip_runtime.h>
#include <stdint.h>

// ---- problem constants (match reference) ----
#define DIN     512
#define HIDDEN  1024
#define DEPTH   64
#define VF      64
#define KSEL    4096
#define RES     256
#define BB      4
#define HH      64
#define WWD     64
#define NPIX    (BB*HH*WWD)        // 16384
#define NVOX    (NPIX*DEPTH)       // 1048576
#define PERBATCH (HH*WWD*DEPTH)    // 262144

typedef __bf16 bf16_t;
typedef __attribute__((ext_vector_type(16))) __bf16 v16bf;
typedef __attribute__((ext_vector_type(8)))  __bf16 v8bf;
typedef __attribute__((ext_vector_type(8)))  float  v8f;
typedef __attribute__((ext_vector_type(4)))  float  v4f;
typedef __attribute__((ext_vector_type(4)))  unsigned u32x4;
typedef __attribute__((ext_vector_type(8)))  int      i32x8;
typedef __attribute__((ext_vector_type(4)))  int      i32x4;

// Tensor Data Mover availability (device pass only; host pass falls back)
#if defined(__AMDGCN__) && __has_builtin(__builtin_amdgcn_tensor_load_to_lds) && __has_builtin(__builtin_amdgcn_s_wait_tensorcnt)
#define USE_TDM 1
#else
#define USE_TDM 0
#endif

__device__ __forceinline__ float gelu_tanh(float x) {
    const float c0 = 0.7978845608028654f; // sqrt(2/pi)
    return 0.5f * x * (1.0f + tanhf(c0 * (x + 0.044715f * x * x * x)));
}
__device__ __forceinline__ float sigmoidf(float x) { return 1.0f / (1.0f + expf(-x)); }

// ---- WMMA fragment loaders (wave32, 16x16x32 bf16) ----
__device__ __forceinline__ v16bf load_a_bf16(const bf16_t* A, int lda, int row, int k0, int half) {
    const bf16_t* base = A + (size_t)row * lda + k0 + 8 * half;
    const v8bf lo = *(const v8bf*)(base);
    const v8bf hi = *(const v8bf*)(base + 16);
    v16bf r;
#pragma unroll
    for (int i = 0; i < 8; ++i) { r[i] = lo[i]; r[i + 8] = hi[i]; }
    return r;
}
__device__ __forceinline__ v16bf load_a_f32(const float* A, int lda, int row, int k0, int half) {
    const float* base = A + (size_t)row * lda + k0 + 8 * half;
    const v4f q0 = *(const v4f*)(base);
    const v4f q1 = *(const v4f*)(base + 4);
    const v4f q2 = *(const v4f*)(base + 16);
    const v4f q3 = *(const v4f*)(base + 20);
    v16bf r;
#pragma unroll
    for (int i = 0; i < 4; ++i) {
        r[i]      = (bf16_t)q0[i];
        r[i + 4]  = (bf16_t)q1[i];
        r[i + 8]  = (bf16_t)q2[i];
        r[i + 12] = (bf16_t)q3[i];
    }
    return r;
}
// B as W^T from global: row-major [N, Kdim] bf16.
__device__ __forceinline__ v16bf load_b_global(const bf16_t* Bt, int ldb, int n, int k0, int half) {
    const bf16_t* base = Bt + (size_t)n * ldb + k0 + 8 * half;
    const v8bf lo = *(const v8bf*)(base);
    const v8bf hi = *(const v8bf*)(base + 16);
    return __builtin_shufflevector(lo, hi, 0,1,2,3,4,5,6,7,8,9,10,11,12,13,14,15);
}
// B from LDS tile: [TN rows][32 K] bf16, row-major.
__device__ __forceinline__ v16bf load_b_lds(const bf16_t* tile, int n, int half) {
    const bf16_t* base = tile + n * 32 + 8 * half;
    const v8bf lo = *(const v8bf*)(base);
    const v8bf hi = *(const v8bf*)(base + 16);
    return __builtin_shufflevector(lo, hi, 0,1,2,3,4,5,6,7,8,9,10,11,12,13,14,15);
}

#if USE_TDM
// Issue TENSOR_LOAD_TO_LDS of a [tileN x 32]-element bf16 tile from W^T[N, Kdim]
// starting at (row n0, col k0) into LDS offset ldsOff. 2D descriptor (groups 2/3 zero).
__device__ __forceinline__ void tdm_load_b_tile(unsigned ldsOff, const bf16_t* gsrc,
                                                int Kdim, int N, int tileN)
{
    const unsigned long long ga = (unsigned long long)(uintptr_t)gsrc;
    u32x4 g0;
    g0[0] = 1u;                                   // count=1 (valid user descriptor)
    g0[1] = ldsOff;                               // lds_addr (bytes)
    g0[2] = (unsigned)(ga & 0xFFFFFFFFull);       // global_addr[31:0]
    g0[3] = (unsigned)((ga >> 32) & 0x01FFFFFFull) | (2u << 30); // addr[56:32] | type=2
    i32x8 g1;
    g1[0] = (int)(1u << 16);                      // workgroup_mask=0 | data_size=1 (2B)
    g1[1] = (int)(((unsigned)Kdim & 0xFFFFu) << 16);              // tensor_dim0 lo16
    g1[2] = (int)((((unsigned)Kdim >> 16) & 0xFFFFu) | (((unsigned)N & 0xFFFFu) << 16));
    g1[3] = (int)((((unsigned)N >> 16) & 0xFFFFu) | (32u << 16)); // tile_dim0 = 32 (K)
    g1[4] = (int)((unsigned)tileN & 0xFFFFu);     // tile_dim1 = tileN, tile_dim2 = 0
    g1[5] = (int)Kdim;                            // tensor_dim0_stride (elements)
    g1[6] = 0;
    g1[7] = 0;
    const i32x4 z4 = {0, 0, 0, 0};
#if defined(__clang_major__) && (__clang_major__ >= 23)
    const i32x8 z8 = {0, 0, 0, 0, 0, 0, 0, 0};
    __builtin_amdgcn_tensor_load_to_lds(g0, g1, z4, z4, z8, 0);
#else
    __builtin_amdgcn_tensor_load_to_lds(g0, g1, z4, z4, 0);
#endif
}
#endif

// ---- generic WMMA GEMM: C = act(A @ B^T + bias) ----
// Block = BMW*BNW waves; wave tile 64x64. B tile (TN x 32) double-buffered in LDS via TDM.
template<int BMW, int BNW, bool AF32, bool ACT_GELU, bool ST_F32, bool ST_BF16>
__global__ __launch_bounds__(32 * BMW * BNW)
void wmma_gemm_kernel(const void* __restrict__ Av, const bf16_t* __restrict__ Bt,
                      const float* __restrict__ bias,
                      float* __restrict__ Cf, bf16_t* __restrict__ Cb,
                      int M, int N, int Kdim)
{
    constexpr int TN = 64 * BNW;
    __shared__ __align__(128) bf16_t smemB[2 * TN * 32];

    const int lane = threadIdx.x & 31;
    const int wave = threadIdx.x >> 5;
    const int wm = wave / BNW, wn = wave % BNW;
    const int half = lane >> 4, lrow = lane & 15;
    const int rowBase  = (blockIdx.y * BMW + wm) * 64;
    const int colBase0 = blockIdx.x * TN;         // block's N origin
    const int nLocal   = wn * 64;                 // wave's N offset inside block tile

    v8f acc[4][4];
#pragma unroll
    for (int mi = 0; mi < 4; ++mi)
#pragma unroll
        for (int ni = 0; ni < 4; ++ni)
            acc[mi][ni] = (v8f){0.f,0.f,0.f,0.f,0.f,0.f,0.f,0.f};

#if USE_TDM
    if (wave == 0)   // prologue: DMA first B tile into buffer 0
        tdm_load_b_tile((unsigned)(uintptr_t)&smemB[0],
                        Bt + (size_t)colBase0 * Kdim, Kdim, N, TN);
#endif

    for (int k0 = 0; k0 < Kdim; k0 += 32) {
        const int buf = (k0 >> 5) & 1;

#if USE_TDM
        if (wave == 0)
            __builtin_amdgcn_s_wait_tensorcnt((short)0);   // tile `buf` landed
        __syncthreads();                                   // visible to all; buf^1 free
        if (wave == 0 && (k0 + 32) < Kdim)                 // overlap next DMA with compute
            tdm_load_b_tile((unsigned)(uintptr_t)&smemB[(buf ^ 1) * TN * 32],
                            Bt + (size_t)colBase0 * Kdim + (k0 + 32), Kdim, N, TN);
        const bf16_t* btile = &smemB[buf * TN * 32];
#endif

        v16bf a[4], b[4];
#pragma unroll
        for (int mi = 0; mi < 4; ++mi) {
            const int row = rowBase + 16 * mi + lrow;
            if constexpr (AF32) a[mi] = load_a_f32((const float*)Av, Kdim, row, k0, half);
            else                a[mi] = load_a_bf16((const bf16_t*)Av, Kdim, row, k0, half);
        }
#pragma unroll
        for (int ni = 0; ni < 4; ++ni) {
#if USE_TDM
            b[ni] = load_b_lds(btile, nLocal + 16 * ni + lrow, half);
#else
            b[ni] = load_b_global(Bt, Kdim, colBase0 + nLocal + 16 * ni + lrow, k0, half);
#endif
        }

        if (k0 + 32 < Kdim) { // prefetch next A rows (global_prefetch_b8)
            if constexpr (AF32)
                __builtin_prefetch((const char*)Av + ((size_t)(rowBase + lrow) * Kdim + k0 + 32) * 4, 0, 1);
            else
                __builtin_prefetch((const char*)Av + ((size_t)(rowBase + lrow) * Kdim + k0 + 32) * 2, 0, 1);
        }

#pragma unroll
        for (int mi = 0; mi < 4; ++mi)
#pragma unroll
            for (int ni = 0; ni < 4; ++ni)
                acc[mi][ni] = __builtin_amdgcn_wmma_f32_16x16x32_bf16(
                    false, a[mi], false, b[ni], (short)0, acc[mi][ni], false, false);
    }

#pragma unroll
    for (int ni = 0; ni < 4; ++ni) {
        const int col = colBase0 + nLocal + 16 * ni + lrow;
        const float bv = bias[col];
#pragma unroll
        for (int mi = 0; mi < 4; ++mi) {
#pragma unroll
            for (int v = 0; v < 8; ++v) {
                const int row = rowBase + 16 * mi + v + 8 * half;
                float x = acc[mi][ni][v] + bv;
                if constexpr (ACT_GELU) x = gelu_tanh(x);
                if constexpr (ST_F32)  Cf[(size_t)row * N + col] = x;
                if constexpr (ST_BF16) Cb[(size_t)row * N + col] = (bf16_t)x;
            }
        }
    }
}

// ---- weight transpose + f32->bf16 convert: out[c*R + r] = in[r*C + c] ----
__global__ void transpose_cvt_kernel(const float* __restrict__ in, bf16_t* __restrict__ out,
                                     int R, int C)
{
    int i = blockIdx.x * blockDim.x + threadIdx.x;
    if (i >= R * C) return;
    int r = i / C, c = i % C;
    out[(size_t)c * R + r] = (bf16_t)in[i];
}

// ---- per-pixel softmax over DEPTH logits ----
__global__ void softmax_kernel(const float* __restrict__ logits, float* __restrict__ probs)
{
    int p = blockIdx.x * blockDim.x + threadIdx.x;
    if (p >= NPIX) return;
    const float* in = logits + (size_t)p * DEPTH;
    float* out = probs + (size_t)p * DEPTH;
    float m = -1e30f;
    for (int i = 0; i < DEPTH; ++i) m = fmaxf(m, in[i]);
    float s = 0.f;
    for (int i = 0; i < DEPTH; ++i) s += expf(in[i] - m);
    float inv = 1.0f / s;
    for (int i = 0; i < DEPTH; ++i) out[i] = expf(in[i] - m) * inv;
}

// ---- fused confidence head: conf = sigmoid(gelu(vox@wc1+bc1)@wc2+bc2) * depth_prob ----
__global__ __launch_bounds__(128)
void conf_kernel(const float* __restrict__ Vox, const bf16_t* __restrict__ Wc1T,
                 const float* __restrict__ bc1, const float* __restrict__ wc2,
                 const float* __restrict__ bc2, const float* __restrict__ dprob,
                 float* __restrict__ conf)
{
    const int lane = threadIdx.x & 31;
    const int wave = threadIdx.x >> 5;
    const int half = lane >> 4, lrow = lane & 15;
    const int rowBase = (blockIdx.x * 4 + wave) * 64;

    v8f acc[4][4];
#pragma unroll
    for (int mi = 0; mi < 4; ++mi)
#pragma unroll
        for (int ni = 0; ni < 4; ++ni)
            acc[mi][ni] = (v8f){0.f,0.f,0.f,0.f,0.f,0.f,0.f,0.f};

    for (int k0 = 0; k0 < VF; k0 += 32) {
        v16bf a[4], b[4];
#pragma unroll
        for (int mi = 0; mi < 4; ++mi)
            a[mi] = load_a_f32(Vox, VF, rowBase + 16 * mi + lrow, k0, half);
#pragma unroll
        for (int ni = 0; ni < 4; ++ni)
            b[ni] = load_b_global(Wc1T, VF, 16 * ni + lrow, k0, half);
#pragma unroll
        for (int mi = 0; mi < 4; ++mi)
#pragma unroll
            for (int ni = 0; ni < 4; ++ni)
                acc[mi][ni] = __builtin_amdgcn_wmma_f32_16x16x32_bf16(
                    false, a[mi], false, b[ni], (short)0, acc[mi][ni], false, false);
    }

    float bco[4], w2v[4];
#pragma unroll
    for (int ni = 0; ni < 4; ++ni) {
        bco[ni] = bc1[16 * ni + lrow];
        w2v[ni] = wc2[16 * ni + lrow];
    }
    const float bias2 = bc2[0];

#pragma unroll
    for (int mi = 0; mi < 4; ++mi) {
#pragma unroll
        for (int v = 0; v < 8; ++v) {
            float p = 0.f;
#pragma unroll
            for (int ni = 0; ni < 4; ++ni)
                p += gelu_tanh(acc[mi][ni][v] + bco[ni]) * w2v[ni];
            p += __shfl_xor(p, 1, 32);
            p += __shfl_xor(p, 2, 32);
            p += __shfl_xor(p, 4, 32);
            p += __shfl_xor(p, 8, 32);
            if (lrow == 0) {
                const int r = rowBase + 16 * mi + v + 8 * half;
                conf[r] = sigmoidf(p + bias2) * dprob[r];
            }
        }
    }
}

// ---- top-K: exact threshold per batch via binary search on float bits ----
__global__ __launch_bounds__(1024)
void topk_thresh_kernel(const float* __restrict__ conf, unsigned* __restrict__ thr,
                        unsigned* __restrict__ ng)
{
    __shared__ unsigned sred[32];
    const int b = blockIdx.x;
    const float* cb = conf + (size_t)b * PERBATCH;
    unsigned lo = 0u, hi = 0xFFFFFFFFu;
    for (int it = 0; it < 33; ++it) {
        if (lo >= hi) break; // uniform across block
        unsigned mid = lo + ((hi - lo) >> 1);
        unsigned cnt = 0;
        for (int i = threadIdx.x; i < PERBATCH; i += 1024)
            cnt += (__float_as_uint(cb[i]) > mid) ? 1u : 0u;
        for (int off = 16; off > 0; off >>= 1) cnt += __shfl_xor(cnt, off, 32);
        if ((threadIdx.x & 31) == 0) sred[threadIdx.x >> 5] = cnt;
        __syncthreads();
        unsigned tot = 0;
        for (int t = 0; t < 32; ++t) tot += sred[t];
        __syncthreads();
        if (tot <= (unsigned)KSEL) hi = mid; else lo = mid + 1;
    }
    const unsigned T = lo;
    unsigned cnt = 0;
    for (int i = threadIdx.x; i < PERBATCH; i += 1024)
        cnt += (__float_as_uint(cb[i]) > T) ? 1u : 0u;
    for (int off = 16; off > 0; off >>= 1) cnt += __shfl_xor(cnt, off, 32);
    if ((threadIdx.x & 31) == 0) sred[threadIdx.x >> 5] = cnt;
    __syncthreads();
    unsigned tot = 0;
    for (int t = 0; t < 32; ++t) tot += sred[t];
    if (threadIdx.x == 0) { thr[b] = T; ng[b] = tot; }
}

__global__ void zero_counters_kernel(unsigned* __restrict__ c, int n)
{
    int i = blockIdx.x * blockDim.x + threadIdx.x;
    if (i < n) c[i] = 0u;
}

__global__ void topk_compact_kernel(const float* __restrict__ conf,
                                    const unsigned* __restrict__ thr,
                                    const unsigned* __restrict__ ng,
                                    unsigned* __restrict__ cntHi, unsigned* __restrict__ cntEq,
                                    int* __restrict__ topkIdx, float* __restrict__ outConf)
{
    int g = blockIdx.x * blockDim.x + threadIdx.x;
    if (g >= BB * PERBATCH) return;
    int b = g / PERBATCH, i = g - b * PERBATCH;
    float c = conf[g];
    unsigned u = __float_as_uint(c);
    unsigned T = thr[b];
    int slot = -1;
    if (u > T) {
        slot = (int)atomicAdd(&cntHi[b], 1u);
    } else if (u == T) {
        unsigned e = atomicAdd(&cntEq[b], 1u);
        unsigned rem = (unsigned)KSEL - ng[b];
        if (e < rem) slot = (int)(ng[b] + e);
    }
    if (slot >= 0) {
        topkIdx[b * KSEL + slot] = i;
        outConf[b * KSEL + slot] = c;
    }
}

// ---- per-selected-voxel heads: coords, op classifier, pbr decoder, tf copy ----
__global__ void heads_kernel(const float* __restrict__ Vox, const int* __restrict__ topkIdx,
                             const float* __restrict__ wo1, const float* __restrict__ bo1,
                             const float* __restrict__ wo2, const float* __restrict__ bo2,
                             const float* __restrict__ wp1, const float* __restrict__ bp1,
                             const float* __restrict__ wp2, const float* __restrict__ bp2,
                             int* __restrict__ coordsOut, float* __restrict__ featsOut,
                             int* __restrict__ opOut, float* __restrict__ logitsOut,
                             float* __restrict__ tfOut)
{
    int j = blockIdx.x * blockDim.x + threadIdx.x;
    if (j >= BB * KSEL) return;
    const int b = j / KSEL;
    const int idx = topkIdx[j];

    const int d  = idx % DEPTH;
    const int hw = idx / DEPTH;
    const int hi = hw / WWD;
    const int wi = hw % WWD;
    coordsOut[4 * j + 0] = b;
    coordsOut[4 * j + 1] = (hi * RES) / HH;
    coordsOut[4 * j + 2] = (wi * RES) / WWD;
    coordsOut[4 * j + 3] = (d * RES) / DEPTH;

    const float* tf = Vox + (size_t)b * ((size_t)PERBATCH * VF) + (size_t)idx * VF;
    float f[VF];
#pragma unroll 8
    for (int i = 0; i < VF; ++i) { f[i] = tf[i]; tfOut[(size_t)j * VF + i] = f[i]; }

    float lg[3] = {bo2[0], bo2[1], bo2[2]};
    for (int jj = 0; jj < 32; ++jj) {
        float h = bo1[jj];
        for (int ff = 0; ff < VF; ++ff) h += f[ff] * wo1[ff * 32 + jj];
        h = gelu_tanh(h);
        lg[0] += h * wo2[jj * 3 + 0];
        lg[1] += h * wo2[jj * 3 + 1];
        lg[2] += h * wo2[jj * 3 + 2];
    }
    logitsOut[3 * j + 0] = lg[0];
    logitsOut[3 * j + 1] = lg[1];
    logitsOut[3 * j + 2] = lg[2];
    int am = 0;
    if (lg[1] > lg[am]) am = 1;
    if (lg[2] > lg[am]) am = 2;
    opOut[j] = am;

    float pb[7];
#pragma unroll
    for (int t = 0; t < 7; ++t) pb[t] = bp2[t];
    for (int jj = 0; jj < 64; ++jj) {
        float h = bp1[jj];
        for (int ff = 0; ff < VF; ++ff) h += f[ff] * wp1[ff * 64 + jj];
        h = gelu_tanh(h);
#pragma unroll
        for (int t = 0; t < 7; ++t) pb[t] += h * wp2[jj * 7 + t];
    }
#pragma unroll
    for (int t = 0; t < 6; ++t) featsOut[7 * j + t] = sigmoidf(pb[t]);
    featsOut[7 * j + 6] = tanhf(pb[6]);
}

static inline size_t align256(size_t x) { return (x + 255) & ~(size_t)255; }

extern "C" void kernel_launch(void* const* d_in, const int* in_sizes, int n_in,
                              void* d_out, int out_size, void* d_ws, size_t ws_size,
                              hipStream_t stream)
{
    (void)in_sizes; (void)n_in; (void)out_size; (void)ws_size;
    const float* features = (const float*)d_in[0];
    const float* w1  = (const float*)d_in[1];
    const float* b1  = (const float*)d_in[2];
    const float* w2  = (const float*)d_in[3];
    const float* b2  = (const float*)d_in[4];
    const float* wd  = (const float*)d_in[5];
    const float* bd  = (const float*)d_in[6];
    const float* wc1 = (const float*)d_in[7];
    const float* bc1 = (const float*)d_in[8];
    const float* wc2 = (const float*)d_in[9];
    const float* bc2 = (const float*)d_in[10];
    const float* wo1 = (const float*)d_in[11];
    const float* bo1 = (const float*)d_in[12];
    const float* wo2 = (const float*)d_in[13];
    const float* bo2 = (const float*)d_in[14];
    const float* wp1 = (const float*)d_in[15];
    const float* bp1 = (const float*)d_in[16];
    const float* wp2 = (const float*)d_in[17];
    const float* bp2 = (const float*)d_in[18];

    char* ws = (char*)d_ws;
    size_t o = 0;
    bf16_t* W1T  = (bf16_t*)(ws + o); o = align256(o + (size_t)HIDDEN * DIN * 2);
    bf16_t* W2T  = (bf16_t*)(ws + o); o = align256(o + (size_t)(DEPTH * VF) * HIDDEN * 2);
    bf16_t* WdT  = (bf16_t*)(ws + o); o = align256(o + (size_t)DEPTH * DIN * 2);
    bf16_t* Wc1T = (bf16_t*)(ws + o); o = align256(o + (size_t)VF * VF * 2);
    bf16_t* Hid  = (bf16_t*)(ws + o); o = align256(o + (size_t)NPIX * HIDDEN * 2);
    float*  Vox  = (float*)(ws + o);  o = align256(o + (size_t)NPIX * DEPTH * VF * 4);
    float*  Dlog = (float*)(ws + o);  o = align256(o + (size_t)NPIX * DEPTH * 4);
    float*  Dprb = (float*)(ws + o);  o = align256(o + (size_t)NPIX * DEPTH * 4);
    float*  Conf = (float*)(ws + o);  o = align256(o + (size_t)NVOX * 4);
    unsigned* Thr   = (unsigned*)(ws + o); o = align256(o + BB * 4);
    unsigned* Ng    = (unsigned*)(ws + o); o = align256(o + BB * 4);
    unsigned* CntHi = (unsigned*)(ws + o); o = align256(o + BB * 4);
    unsigned* CntEq = (unsigned*)(ws + o); o = align256(o + BB * 4);
    int* TopkIdx    = (int*)(ws + o);      o = align256(o + (size_t)BB * KSEL * 4);

    float* out = (float*)d_out;
    int*   coordsOut = (int*)out;
    float* featsOut  = out + (size_t)BB * KSEL * 4;
    int*   opOut     = (int*)(out + (size_t)BB * KSEL * 11);
    float* confOut   = out + (size_t)BB * KSEL * 12;
    float* logitsOut = out + (size_t)BB * KSEL * 13;
    float* tfOut     = out + (size_t)BB * KSEL * 16;

    // 1) weight transpose + bf16 convert
    {
        int t1 = DIN * HIDDEN;
        transpose_cvt_kernel<<<(t1 + 255) / 256, 256, 0, stream>>>(w1, W1T, DIN, HIDDEN);
        int t2 = HIDDEN * DEPTH * VF;
        transpose_cvt_kernel<<<(t2 + 255) / 256, 256, 0, stream>>>(w2, W2T, HIDDEN, DEPTH * VF);
        int t3 = DIN * DEPTH;
        transpose_cvt_kernel<<<(t3 + 255) / 256, 256, 0, stream>>>(wd, WdT, DIN, DEPTH);
        int t4 = VF * VF;
        transpose_cvt_kernel<<<(t4 + 255) / 256, 256, 0, stream>>>(wc1, Wc1T, VF, VF);
    }

    // 2) GEMM1: hid = gelu(features @ w1 + b1) -> bf16, [16384 x 1024], K=512
    wmma_gemm_kernel<2, 2, true, true, false, true>
        <<<dim3(HIDDEN / 128, NPIX / 128), 128, 0, stream>>>(
            features, W1T, b1, (float*)nullptr, Hid, NPIX, HIDDEN, DIN);

    // 3) GEMM2: vox = hid @ w2 + b2 -> f32, [16384 x 4096], K=1024 (dominant)
    wmma_gemm_kernel<2, 2, false, false, true, false>
        <<<dim3((DEPTH * VF) / 128, NPIX / 128), 128, 0, stream>>>(
            Hid, W2T, b2, Vox, (bf16_t*)nullptr, NPIX, DEPTH * VF, HIDDEN);

    // 4) depth logits: features @ wd + bd -> f32, [16384 x 64], K=512
    wmma_gemm_kernel<4, 1, true, false, true, false>
        <<<dim3(1, NPIX / 256), 128, 0, stream>>>(
            features, WdT, bd, Dlog, (bf16_t*)nullptr, NPIX, DEPTH, DIN);

    // 5) depth softmax
    softmax_kernel<<<(NPIX + 255) / 256, 256, 0, stream>>>(Dlog, Dprb);

    // 6) fused confidence head (WMMA) + depth_prob multiply
    conf_kernel<<<NVOX / 256, 128, 0, stream>>>(Vox, Wc1T, bc1, wc2, bc2, Dprb, Conf);

    // 7) top-K per batch
    zero_counters_kernel<<<1, 32, 0, stream>>>(CntHi, BB);
    zero_counters_kernel<<<1, 32, 0, stream>>>(CntEq, BB);
    topk_thresh_kernel<<<BB, 1024, 0, stream>>>(Conf, Thr, Ng);
    topk_compact_kernel<<<(BB * PERBATCH + 255) / 256, 256, 0, stream>>>(
        Conf, Thr, Ng, CntHi, CntEq, TopkIdx, confOut);

    // 8) per-voxel heads + outputs
    heads_kernel<<<(BB * KSEL + 255) / 256, 256, 0, stream>>>(
        Vox, TopkIdx, wo1, bo1, wo2, bo2, wp1, bp1, wp2, bp2,
        coordsOut, featsOut, opOut, logitsOut, tfOut);
}
